// HybridSymbolicTransformerModel_66314295050785
// MI455X (gfx1250) — matrix operand order repaired and checked by reading, not compiled, not run on hardware
//
#include <hip/hip_runtime.h>
#include <hip/hip_bf16.h>

// ---------------------------------------------------------------------------
// Model dims (compile-time constants of this workload)
//   C=1024, H=16, D=64, L=2, V=32000, FF=4096, B=1, T=2048
// ---------------------------------------------------------------------------

typedef __attribute__((ext_vector_type(16))) __bf16 v16bf;
typedef __attribute__((ext_vector_type(8)))  __bf16 v8bf;
typedef __attribute__((ext_vector_type(8)))  float  v8f;

// ---------------------------------------------------------------------------
// WMMA wrapper: D = A(16x32 bf16) x B(32x16 bf16) + C(16x16 f32)
// ---------------------------------------------------------------------------
__device__ inline v8f wmma_bf16(v16bf a, v16bf b, v8f c) {
    return __builtin_amdgcn_wmma_f32_16x16x32_bf16(
        /*neg_a=*/false, a, /*neg_b=*/false, b,
        /*c_mod=*/(short)0, c, /*reuse_a=*/false, /*reuse_b=*/false);
}

// ---------------------------------------------------------------------------
// Fragment loaders (layouts per CDNA5 ISA 7.12.2, wave32)
// A 16x32 (MxK): lanes 0-15 hold M=lane, K in {0..7,16..23}; lanes 16-31 hold
//                M=lane-16, K in {8..15,24..31}.
// B 32x16 (KxN): lanes 0-15 hold N=lane, K in {0..7,16..23}; lanes 16-31 N=lane-16, K in {8..15,24..31}.
// C/D 16x16:     VGPR r: lanes 0-15 -> M=r; lanes 16-31 -> M=8+r; N=lane&15.
// ---------------------------------------------------------------------------

// A fragment from f32 row-major source (convert to bf16 in-register).
__device__ inline v16bf load_A_f32(const float* __restrict__ A, int lda,
                                   int m0, int k0, int lane) {
    const float* p = A + (size_t)(m0 + (lane & 15)) * lda + k0 + ((lane >> 4) << 3);
    float4 x0 = *(const float4*)(p);
    float4 x1 = *(const float4*)(p + 4);
    float4 x2 = *(const float4*)(p + 16);
    float4 x3 = *(const float4*)(p + 20);
    v16bf f;
    f[0]=(__bf16)x0.x;  f[1]=(__bf16)x0.y;  f[2]=(__bf16)x0.z;  f[3]=(__bf16)x0.w;
    f[4]=(__bf16)x1.x;  f[5]=(__bf16)x1.y;  f[6]=(__bf16)x1.z;  f[7]=(__bf16)x1.w;
    f[8]=(__bf16)x2.x;  f[9]=(__bf16)x2.y;  f[10]=(__bf16)x2.z; f[11]=(__bf16)x2.w;
    f[12]=(__bf16)x3.x; f[13]=(__bf16)x3.y; f[14]=(__bf16)x3.z; f[15]=(__bf16)x3.w;
    return f;
}

// B fragment from bf16 source stored row-major as (N x K): B[k][n] = W[n][k].
__device__ inline v16bf load_B_bf16(const __bf16* __restrict__ W, int ldw,
                                    int n0, int k0, int lane) {
    const __bf16* p = W + (size_t)(n0 + (lane & 15)) * ldw + k0 + ((lane >> 4) << 3);
    v8bf lo = *(const v8bf*)(p);
    v8bf hi = *(const v8bf*)(p + 16);
    v16bf f;
#pragma unroll
    for (int e = 0; e < 8; ++e) { f[e] = lo[e]; f[8 + e] = hi[e]; }
    return f;
}

// B fragment from f32 source stored row-major as (N x K) (convert to bf16).
__device__ inline v16bf load_B_f32(const float* __restrict__ W, int ldw,
                                   int n0, int k0, int lane) {
    const float* p = W + (size_t)(n0 + (lane & 15)) * ldw + k0 + ((lane >> 4) << 3);
    float4 x0 = *(const float4*)(p);
    float4 x1 = *(const float4*)(p + 4);
    float4 x2 = *(const float4*)(p + 16);
    float4 x3 = *(const float4*)(p + 20);
    v16bf f;
    f[0]=(__bf16)x0.x;  f[1]=(__bf16)x0.y;  f[2]=(__bf16)x0.z;  f[3]=(__bf16)x0.w;
    f[4]=(__bf16)x1.x;  f[5]=(__bf16)x1.y;  f[6]=(__bf16)x1.z;  f[7]=(__bf16)x1.w;
    f[8]=(__bf16)x2.x;  f[9]=(__bf16)x2.y;  f[10]=(__bf16)x2.z; f[11]=(__bf16)x2.w;
    f[12]=(__bf16)x3.x; f[13]=(__bf16)x3.y; f[14]=(__bf16)x3.z; f[15]=(__bf16)x3.w;
    return f;
}

// ---------------------------------------------------------------------------
// Generic WMMA GEMM:  Out(MxN) = A(MxK,f32) * W(NxK,bf16)^T [+bias] [GELU] [+=]
// Block: 256 threads = 8 waves, wave tile 32x64 (2x4 WMMA accumulators),
// waves arranged 4(M) x 2(N) -> block tile 128x128.
// blockIdx.z applies head offsets (for block-diagonal per-head matmuls).
// flags: bit0 = accumulate into Out (+=), bit1 = exact GELU epilogue.
// ---------------------------------------------------------------------------
__device__ inline void store_tile(float* __restrict__ Out, int ldo,
                                  const float* __restrict__ bias, v8f acc,
                                  int m0, int n0, int lane, int flags) {
    int nn = n0 + (lane & 15);
    int rb = m0 + ((lane >> 4) << 3);
    float bv = bias ? bias[nn] : 0.0f;
#pragma unroll
    for (int r = 0; r < 8; ++r) {
        float v = acc[r] + bv;
        if (flags & 2) v = 0.5f * v * (1.0f + erff(v * 0.70710678118654752f));
        float* o = Out + (size_t)(rb + r) * ldo + nn;
        if (flags & 1) *o += v; else *o = v;
    }
}

__global__ __launch_bounds__(256) void gemm_kernel(
    const float* __restrict__ A, int lda, int zA,
    const __bf16* __restrict__ W, int ldw, int zW,
    const float* __restrict__ bias,
    float* __restrict__ Out, int ldo, int zO,
    int M, int N, int K, int flags) {
    A   += (size_t)blockIdx.z * zA;
    W   += (size_t)blockIdx.z * zW;
    Out += (size_t)blockIdx.z * zO;
    int lane = threadIdx.x & 31;
    int wid  = threadIdx.x >> 5;
    int m0 = blockIdx.x * 128 + (wid & 3) * 32;
    int n0 = blockIdx.y * 128 + (wid >> 2) * 64;
    if (m0 >= M || n0 >= N) return;   // wave-uniform

    v8f acc0[4] = {{}, {}, {}, {}};
    v8f acc1[4] = {{}, {}, {}, {}};
    for (int k0 = 0; k0 < K; k0 += 32) {
        // prefetch next K-step of the streaming weight operand into cache
        if (k0 + 32 < K) {
            __builtin_prefetch(W + (size_t)(n0 + (lane & 15)) * ldw + k0 + 32, 0, 1);
            __builtin_prefetch(W + (size_t)(n0 + 32 + (lane & 15)) * ldw + k0 + 32, 0, 1);
        }
        v16bf a0 = load_A_f32(A, lda, m0,      k0, lane);
        v16bf a1 = load_A_f32(A, lda, m0 + 16, k0, lane);
#pragma unroll
        for (int j = 0; j < 4; ++j) {
            v16bf bj = load_B_bf16(W, ldw, n0 + 16 * j, k0, lane);
            acc0[j] = wmma_bf16(a0, bj, acc0[j]);
            acc1[j] = wmma_bf16(a1, bj, acc1[j]);
        }
    }
#pragma unroll
    for (int j = 0; j < 4; ++j) {
        store_tile(Out, ldo, bias, acc0[j], m0,      n0 + 16 * j, lane, flags);
        store_tile(Out, ldo, bias, acc1[j], m0 + 16, n0 + 16 * j, lane, flags);
    }
}

// ---------------------------------------------------------------------------
// Flash-style attention with ALiBi (no causal mask, matches reference).
// One wave handles one (head, 16-query) tile; streams 32-key chunks.
// qk: (T, 2C) f32 with q at cols [0,C) and k at cols [C,2C).
// vt: (H, D, T) bf16 (pre-transposed V). y: (T, C) f32 output.
// ---------------------------------------------------------------------------
__global__ __launch_bounds__(256) void attn_kernel(
    const float* __restrict__ qk, const __bf16* __restrict__ vt,
    float* __restrict__ y, int T_, int C_, int ld) {
    int h    = blockIdx.y;
    int lane = threadIdx.x & 31;
    int wid  = threadIdx.x >> 5;
    int q0   = (blockIdx.x * 8 + wid) * 16;
    if (q0 >= T_) return;

    const float* q    = qk + h * 64;
    const float* kmat = qk + C_ + h * 64;
    const float  scale = 0.125f;                        // 1/sqrt(64)
    const float  slope = exp2f(-0.5f * (float)(h + 1)); // ALiBi slopes, H=16

    __shared__ __bf16 plds[8][16][32];                  // per-wave P staging

    v16bf qa0 = load_A_f32(q, ld, q0, 0,  lane);
    v16bf qa1 = load_A_f32(q, ld, q0, 32, lane);

    float mrun[8], lrun[8];
#pragma unroll
    for (int r = 0; r < 8; ++r) { mrun[r] = -1e30f; lrun[r] = 0.0f; }
    v8f O0 = {}, O1 = {}, O2 = {}, O3 = {};

    int halfoff = (lane >> 4) << 3;   // 0 or 8 (row offset of this lane half)
    int nl      = lane & 15;

    for (int kc = 0; kc < T_; kc += 32) {
        // S = Q K^T for 2 subtiles of 16 keys
        v8f S0 = {}, S1 = {};
        {
            v16bf b0 = load_B_f32(kmat, ld, kc,      0,  lane);
            v16bf b1 = load_B_f32(kmat, ld, kc,      32, lane);
            S0 = wmma_bf16(qa0, b0, S0);
            S0 = wmma_bf16(qa1, b1, S0);
            v16bf c0 = load_B_f32(kmat, ld, kc + 16, 0,  lane);
            v16bf c1 = load_B_f32(kmat, ld, kc + 16, 32, lane);
            S1 = wmma_bf16(qa0, c0, S1);
            S1 = wmma_bf16(qa1, c1, S1);
        }
        // scale + ALiBi  (bias = slope * min(j - i, 0))
#pragma unroll
        for (int r = 0; r < 8; ++r) {
            float i  = (float)(q0 + halfoff + r);
            float j0 = (float)(kc + nl);
            float j1 = (float)(kc + 16 + nl);
            S0[r] = S0[r] * scale + slope * fminf(j0 - i, 0.0f);
            S1[r] = S1[r] * scale + slope * fminf(j1 - i, 0.0f);
        }
        // chunk row-max (reduce across the 16 lanes of each half)
        float mc[8];
#pragma unroll
        for (int r = 0; r < 8; ++r) mc[r] = fmaxf(S0[r], S1[r]);
#pragma unroll
        for (int off = 1; off < 16; off <<= 1)
#pragma unroll
            for (int r = 0; r < 8; ++r)
                mc[r] = fmaxf(mc[r], __shfl_xor(mc[r], off, 32));

        // online-softmax rescale + P into LDS (bf16)
        float psum[8];
#pragma unroll
        for (int r = 0; r < 8; ++r) {
            float mn = fmaxf(mrun[r], mc[r]);
            float al = __expf(mrun[r] - mn);
            mrun[r] = mn;
            lrun[r] *= al;
            O0[r] *= al; O1[r] *= al; O2[r] *= al; O3[r] *= al;
            float p0 = __expf(S0[r] - mn);
            float p1 = __expf(S1[r] - mn);
            psum[r] = p0 + p1;
            plds[wid][halfoff + r][nl]      = (__bf16)p0;
            plds[wid][halfoff + r][16 + nl] = (__bf16)p1;
        }
#pragma unroll
        for (int off = 1; off < 16; off <<= 1)
#pragma unroll
            for (int r = 0; r < 8; ++r)
                psum[r] += __shfl_xor(psum[r], off, 32);
#pragma unroll
        for (int r = 0; r < 8; ++r) lrun[r] += psum[r];

        asm volatile("s_wait_dscnt 0" ::: "memory");   // LDS stores visible to re-layout read

        // re-read P as an A fragment (16x32, K = 32 keys)
        v16bf pf;
        {
            int pk = (lane >> 4) << 3;
#pragma unroll
            for (int e = 0; e < 8; ++e) {
                pf[e]     = plds[wid][nl][pk + e];
                pf[8 + e] = plds[wid][nl][16 + pk + e];
            }
        }
        // O += P * V   (V^T layout (H,D,T): contiguous keys per dim row)
        const __bf16* vth = vt + (size_t)h * 64 * T_;
        v16bf vb0 = load_B_bf16(vth, T_, 0,  kc, lane);
        v16bf vb1 = load_B_bf16(vth, T_, 16, kc, lane);
        v16bf vb2 = load_B_bf16(vth, T_, 32, kc, lane);
        v16bf vb3 = load_B_bf16(vth, T_, 48, kc, lane);
        O0 = wmma_bf16(pf, vb0, O0);
        O1 = wmma_bf16(pf, vb1, O1);
        O2 = wmma_bf16(pf, vb2, O2);
        O3 = wmma_bf16(pf, vb3, O3);
    }

    // finalize: y = O / l
#pragma unroll
    for (int r = 0; r < 8; ++r) {
        float inv = 1.0f / lrun[r];
        int row = q0 + halfoff + r;
        float* o = y + (size_t)row * C_ + h * 64 + nl;
        o[0]  = O0[r] * inv;
        o[16] = O1[r] * inv;
        o[32] = O2[r] * inv;
        o[48] = O3[r] * inv;
    }
}

// ---------------------------------------------------------------------------
// Per-head LayerNorm over D=64 (scalar scale/bias per head). One wave/item.
// ---------------------------------------------------------------------------
__global__ __launch_bounds__(256) void symln_kernel(
    const float* __restrict__ x, const float* __restrict__ w,
    const float* __restrict__ b, float* __restrict__ out,
    int items, int H_, int C_) {
    int item = blockIdx.x * 8 + (threadIdx.x >> 5);
    if (item >= items) return;
    int lane = threadIdx.x & 31;
    int tok = item / H_, hh = item - tok * H_;
    const float* p = x + (size_t)tok * C_ + hh * 64;
    float a0 = p[lane], a1 = p[lane + 32];
    float s = a0 + a1;
#pragma unroll
    for (int off = 1; off < 32; off <<= 1) s += __shfl_xor(s, off, 32);
    float mu = s * (1.0f / 64.0f);
    float d0 = a0 - mu, d1 = a1 - mu;
    float v = d0 * d0 + d1 * d1;
#pragma unroll
    for (int off = 1; off < 32; off <<= 1) v += __shfl_xor(v, off, 32);
    float rstd = rsqrtf(v * (1.0f / 64.0f) + 1e-5f);
    float ww = w[hh], bb = b[hh];
    float* o = out + (size_t)tok * C_ + hh * 64;
    o[lane]      = d0 * rstd * ww + bb;
    o[lane + 32] = d1 * rstd * ww + bb;
}

// ---------------------------------------------------------------------------
// Small utility kernels
// ---------------------------------------------------------------------------
__global__ void cvt_bf16_kernel(const float* __restrict__ src,
                                __bf16* __restrict__ dst, size_t n) {
    for (size_t i = (size_t)blockIdx.x * blockDim.x + threadIdx.x; i < n;
         i += (size_t)gridDim.x * blockDim.x)
        dst[i] = (__bf16)src[i];
}

__global__ void embed_kernel(const int* __restrict__ idx,
                             const float* __restrict__ wte,
                             float* __restrict__ x, int n, int C_) {
    for (int i = blockIdx.x * blockDim.x + threadIdx.x; i < n;
         i += gridDim.x * blockDim.x) {
        int t = i / C_, c = i - t * C_;
        x[i] = wte[(size_t)idx[t] * C_ + c];
    }
}

// v (T,C) f32 -> vt (H,D,T) = (C,T) bf16
__global__ void vtrans_kernel(const float* __restrict__ v,
                              __bf16* __restrict__ vt, int T_, int C_) {
    for (int i = blockIdx.x * blockDim.x + threadIdx.x; i < T_ * C_;
         i += gridDim.x * blockDim.x) {
        int t = i / C_, c = i - t * C_;
        vt[(size_t)c * T_ + t] = (__bf16)v[i];
    }
}

// ---------------------------------------------------------------------------
// Host-side orchestration
// ---------------------------------------------------------------------------
extern "C" void kernel_launch(void* const* d_in, const int* in_sizes, int n_in,
                              void* d_out, int out_size, void* d_ws, size_t ws_size,
                              hipStream_t stream) {
    (void)in_sizes; (void)n_in; (void)out_size; (void)ws_size;
    const int Cc = 1024, Hh = 16, Dd = 64, Ll = 2, Vv = 32000, FFf = 4096, Tt = 2048;

    const int*   idx    = (const int*)  d_in[0];
    const float* wte    = (const float*)d_in[1];
    const float* ln1w   = (const float*)d_in[2];
    const float* ln1b   = (const float*)d_in[3];
    const float* ln2w   = (const float*)d_in[4];
    const float* ln2b   = (const float*)d_in[5];
    const float* lnfw   = (const float*)d_in[6];
    const float* lnfb   = (const float*)d_in[7];
    const float* attn_w = (const float*)d_in[8];
    const float* attn_b = (const float*)d_in[9];
    const float* v_tmp  = (const float*)d_in[10];
    const float* p_tmp  = (const float*)d_in[11];
    const float* fc_w   = (const float*)d_in[12];
    const float* fc_b   = (const float*)d_in[13];
    const float* pw     = (const float*)d_in[14];
    const float* pb     = (const float*)d_in[15];

    // workspace carve (256B aligned regions)
    char* wp = (char*)d_ws;
    auto carve = [&](size_t bytes) -> void* {
        void* p = (void*)wp;
        wp += (bytes + 255) & ~(size_t)255;
        return p;
    };
    const size_t nAttnW = (size_t)Ll * 2 * Cc * Cc;
    const size_t nHead  = (size_t)Ll * Hh * Dd * Dd;
    const size_t nFc    = (size_t)Ll * FFf * Cc;
    const size_t nWte   = (size_t)Vv * Cc;
    const size_t TC     = (size_t)Tt * Cc;

    __bf16* attnw_bf = (__bf16*)carve(nAttnW * 2);
    __bf16* vtmp_bf  = (__bf16*)carve(nHead * 2);
    __bf16* ptmp_bf  = (__bf16*)carve(nHead * 2);
    __bf16* fcw_bf   = (__bf16*)carve(nFc * 2);
    __bf16* pw_bf    = (__bf16*)carve(nFc * 2);
    __bf16* wte_bf   = (__bf16*)carve(nWte * 2);
    __bf16* vt_bf    = (__bf16*)carve(TC * 2);
    float*  x        = (float*) carve(TC * 4);
    float*  hbuf     = (float*) carve(TC * 4);
    float*  qkbuf    = (float*) carve(TC * 2 * 4);
    float*  vbuf     = (float*) carve(TC * 4);
    float*  ybuf     = (float*) carve(TC * 4);
    float*  ffbuf    = (float*) carve((size_t)Tt * FFf * 4);

    // weights -> bf16 (deterministic each call)
    cvt_bf16_kernel<<<4096, 256, 0, stream>>>(attn_w, attnw_bf, nAttnW);
    cvt_bf16_kernel<<<1024, 256, 0, stream>>>(v_tmp,  vtmp_bf,  nHead);
    cvt_bf16_kernel<<<1024, 256, 0, stream>>>(p_tmp,  ptmp_bf,  nHead);
    cvt_bf16_kernel<<<4096, 256, 0, stream>>>(fc_w,   fcw_bf,   nFc);
    cvt_bf16_kernel<<<4096, 256, 0, stream>>>(pw,     pw_bf,    nFc);
    cvt_bf16_kernel<<<4096, 256, 0, stream>>>(wte,    wte_bf,   nWte);

    // embedding
    embed_kernel<<<2048, 256, 0, stream>>>(idx, wte, x, Tt * Cc, Cc);

    const int lnBlocks = (Tt * Hh) / 8;
    for (int l = 0; l < Ll; ++l) {
        // h = symLN(x; ln1)
        symln_kernel<<<lnBlocks, 256, 0, stream>>>(x, ln1w + l * Hh, ln1b + l * Hh,
                                                   hbuf, Tt * Hh, Hh, Cc);
        // qk = h @ attn_w^T + attn_b        (M=2048, N=2048, K=1024)
        gemm_kernel<<<dim3(Tt / 128, (2 * Cc) / 128, 1), 256, 0, stream>>>(
            hbuf, Cc, 0, attnw_bf + (size_t)l * 2 * Cc * Cc, Cc, 0,
            attn_b + (size_t)l * 2 * Cc, qkbuf, 2 * Cc, 0, Tt, 2 * Cc, Cc, 0);
        // v[:, h] = h[:, h] @ v_tmp[h]^T    (per-head 64x64, z = head)
        gemm_kernel<<<dim3(Tt / 128, 1, Hh), 256, 0, stream>>>(
            hbuf, Cc, Dd, vtmp_bf + (size_t)l * Hh * Dd * Dd, Dd, Dd * Dd,
            nullptr, vbuf, Cc, Dd, Tt, Dd, Dd, 0);
        // V^T staging (H,D,T) bf16
        vtrans_kernel<<<2048, 256, 0, stream>>>(vbuf, vt_bf, Tt, Cc);
        // y = softmax(q k^T / sqrt(D) + alibi) v
        attn_kernel<<<dim3(Tt / 128, Hh), 256, 0, stream>>>(qkbuf, vt_bf, ybuf,
                                                            Tt, Cc, 2 * Cc);
        // x += per-head proj(y)
        gemm_kernel<<<dim3(Tt / 128, 1, Hh), 256, 0, stream>>>(
            ybuf, Cc, Dd, ptmp_bf + (size_t)l * Hh * Dd * Dd, Dd, Dd * Dd,
            nullptr, x, Cc, Dd, Tt, Dd, Dd, /*ADD*/1);
        // h = symLN(x; ln2)
        symln_kernel<<<lnBlocks, 256, 0, stream>>>(x, ln2w + l * Hh, ln2b + l * Hh,
                                                   hbuf, Tt * Hh, Hh, Cc);
        // ff = GELU(h @ fc_w^T + fc_b)      (N=4096)
        gemm_kernel<<<dim3(Tt / 128, FFf / 128, 1), 256, 0, stream>>>(
            hbuf, Cc, 0, fcw_bf + (size_t)l * FFf * Cc, Cc, 0,
            fc_b + (size_t)l * FFf, ffbuf, FFf, 0, Tt, FFf, Cc, /*GELU*/2);
        // x += ff @ proj_w^T + proj_b       (K=4096)
        gemm_kernel<<<dim3(Tt / 128, Cc / 128, 1), 256, 0, stream>>>(
            ffbuf, FFf, 0, pw_bf + (size_t)l * Cc * FFf, FFf, 0,
            pb + (size_t)l * Cc, x, Cc, 0, Tt, Cc, FFf, /*ADD*/1);
    }

    // final LN + tied lm_head:  logits = symLN(x) @ wte^T  (N=32000 = 250*128)
    symln_kernel<<<lnBlocks, 256, 0, stream>>>(x, lnfw, lnfb, hbuf, Tt * Hh, Hh, Cc);
    gemm_kernel<<<dim3(Tt / 128, Vv / 128, 1), 256, 0, stream>>>(
        hbuf, Cc, 0, wte_bf, Cc, 0, nullptr, (float*)d_out, Vv, 0, Tt, Vv, Cc, 0);
}
